// LSTM_85091892068707
// MI455X (gfx1250) — matrix operand order repaired and checked by reading, not compiled, not run on hardware
//
#include <hip/hip_runtime.h>

// ---------------- problem constants ----------------
constexpr int kBatch = 128;   // B  (GEMM M dimension)
constexpr int kSeq   = 1024;  // T
constexpr int kDin   = 128;   // D
constexpr int kHid   = 256;   // H
constexpr int kKTot  = 384;   // D + H  (fused K)
constexpr int kN4H   = 1024;  // 4*H    (gate columns)
constexpr int kCls   = 10;
constexpr int kNWG   = 16;    // persistent workgroups (one h-column tile each)
constexpr int kPad   = 8;     // LDS row padding (bf16 elems) to avoid bank conflicts

typedef float    v8f   __attribute__((ext_vector_type(8)));
typedef __bf16   v16bf __attribute__((ext_vector_type(16)));
typedef unsigned u32x4 __attribute__((ext_vector_type(4)));

union FragAB {
    v16bf  v;      // WMMA operand (16 bf16 per lane)
    __bf16 e[16];
    u32x4  q[2];   // two 16-byte chunks
};

// fast device activations: v_exp_f32 + v_rcp_f32 (results feed bf16 state)
__device__ __forceinline__ float sigm_fast(float x) {
    return __builtin_amdgcn_rcpf(1.0f + __expf(-x));
}
__device__ __forceinline__ float tanh_fast(float x) {
    const float e2 = __expf(2.0f * x);
    return (e2 - 1.0f) * __builtin_amdgcn_rcpf(e2 + 1.0f);
}

// ---------------------------------------------------------------------------
// Kernel 1: pack weights into bf16 W^T [N=1024][K=384], pack bias [1024] f32,
//           zero h0 (bf16) and the grid-barrier counter.  Runs every call so
//           the persistent kernel's barrier state is deterministic.
// ---------------------------------------------------------------------------
__global__ void lstm_pack(const float* __restrict__ Wgx, const float* __restrict__ Wgh, const float* __restrict__ bg,
                          const float* __restrict__ Wix, const float* __restrict__ Wih, const float* __restrict__ bi,
                          const float* __restrict__ Wfx, const float* __restrict__ Wfh, const float* __restrict__ bfp,
                          const float* __restrict__ Wox, const float* __restrict__ Woh, const float* __restrict__ bo,
                          __bf16* __restrict__ Wt, float* __restrict__ bias,
                          __bf16* __restrict__ hbuf, unsigned* __restrict__ cnt)
{
    const int n = blockIdx.x;        // gate column 0..1023  (order g,i,f,o)
    const int k = threadIdx.x;       // fused K index 0..383
    const int gate = n >> 8;
    const int nh   = n & 255;

    const float* Wx[4] = {Wgx, Wix, Wfx, Wox};
    const float* Wh[4] = {Wgh, Wih, Wfh, Woh};
    const float* bb[4] = {bg,  bi,  bfp, bo };

    float v = (k < kDin) ? Wx[gate][k * kHid + nh]
                         : Wh[gate][(k - kDin) * kHid + nh];
    Wt[(size_t)n * kKTot + k] = (__bf16)v;

    if (k == 0) bias[n] = bb[gate][nh];
    if (n < kBatch && k < kHid) hbuf[n * kHid + k] = (__bf16)0.0f;  // h0 = 0
    if (n == 0 && k == 0) *cnt = 0u;                                 // barrier reset
}

// ---------------------------------------------------------------------------
// Kernel 2: persistent recurrent kernel.
//   grid = 16 WGs x 256 threads (8 wave32 waves).
//   WG j owns h columns [16j, 16j+16); wave w owns batch rows [16w, 16w+16).
//   c-state lives in VGPRs (v8f per wave) for all 1024 steps.
//   Weights for this WG's 4 gate tiles live in LDS (~49KB, loaded once).
// ---------------------------------------------------------------------------
__launch_bounds__(256, 1)
__global__ void lstm_rec(const float*  __restrict__ x,     // [B,T,D] f32
                         const __bf16* __restrict__ Wt,    // [1024][384] bf16 (N-major)
                         const float*  __restrict__ bias,  // [1024] f32
                         __bf16* __restrict__ hbuf,        // [B,H] bf16 (broadcast state)
                         float*  __restrict__ hfin,        // [B,H] f32 (final h)
                         unsigned* __restrict__ cnt)       // grid barrier counter
{
    __shared__ __align__(32) __bf16 sW[64][kKTot + kPad];  // 4 gates x 16 cols, K-major rows
    __shared__ float sB[64];

    const int tid  = threadIdx.x;
    const int j    = blockIdx.x;      // h column tile 0..15
    const int w    = tid >> 5;        // wave id -> M tile
    const int lane = tid & 31;
    const int hl   = lane >> 4;       // lane half (ISA fragment layout)
    const int ll   = lane & 15;

    // ---- stage this WG's weight slice into LDS (once; reused 1024 steps) ----
    for (int idx = tid; idx < 64 * kKTot; idx += 256) {
        const int r = idx / kKTot;
        const int k = idx - r * kKTot;
        const int nrow = (r >> 4) * kHid + j * 16 + (r & 15);   // gate*256 + col
        sW[r][k] = Wt[(size_t)nrow * kKTot + k];
    }
    if (tid < 64) sB[tid] = bias[(tid >> 4) * kHid + j * 16 + (tid & 15)];
    __syncthreads();

    const int m0 = w * 16;
    const int mr = m0 + ll;           // this lane's A-matrix row (batch index)
    const float bb0 = sB[ 0 + ll];
    const float bb1 = sB[16 + ll];
    const float bb2 = sB[32 + ll];
    const float bb3 = sB[48 + ll];

    float cfr[8];
    #pragma unroll
    for (int r = 0; r < 8; ++r) cfr[r] = 0.0f;

    const v8f vzero = {0.f,0.f,0.f,0.f,0.f,0.f,0.f,0.f};

    #pragma unroll 1
    for (int t = 0; t < kSeq; ++t) {
        const float*  xrow = x    + ((size_t)mr * kSeq + t) * kDin;
        const __bf16* hrow = hbuf + mr * kHid;

        // speculative prefetch of next timestep's x row -> global_prefetch_b8
        const int tn = (t + 1 < kSeq) ? (t + 1) : t;
        __builtin_prefetch(x + ((size_t)mr * kSeq + tn) * kDin, 0, 1);

        v8f acc[4];
        acc[0] = vzero; acc[1] = vzero; acc[2] = vzero; acc[3] = vzero;

        // A-fragment loader: two contiguous 16B K-chunks per lane
        auto loadA = [&](int kc, FragAB& a) {
            const int k0 = kc * 32 + hl * 8;   // first chunk; second at +16
            if (kc < 4) {
                // x region: fp32 -> bf16 convert (co-executes with WMMA)
                const float* p0 = xrow + k0;
                #pragma unroll
                for (int i = 0; i < 8; ++i) {
                    a.e[i]     = (__bf16)p0[i];
                    a.e[i + 8] = (__bf16)p0[i + 16];
                }
            } else {
                const u32x4* p = (const u32x4*)(hrow + (k0 - kDin));
                a.q[0] = p[0];
                a.q[1] = p[2];   // +16 elements = +32 bytes
            }
        };

        FragAB a, an;
        loadA(0, a);

        // gates[16x64] = [x_t | h] (16x384) @ W (384x64), K tiled by 32.
        // Batch the 4 gate B-fragment LDS loads (one DS clause + single wait),
        // then issue 4 back-to-back WMMAs; pipeline the next A fragment.
        #pragma unroll
        for (int kc = 0; kc < 12; ++kc) {
            FragAB bfr[4];
            #pragma unroll
            for (int g = 0; g < 4; ++g) {
                const u32x4* pb =
                    (const u32x4*)&sW[g * 16 + ll][kc * 32 + hl * 8];
                bfr[g].q[0] = pb[0];
                bfr[g].q[1] = pb[2];
            }
            if (kc < 11) loadA(kc + 1, an);
            #pragma unroll
            for (int g = 0; g < 4; ++g) {
                acc[g] = __builtin_amdgcn_wmma_f32_16x16x32_bf16(
                            false, a.v, false, bfr[g].v, (short)0, acc[g],
                            false, false);
            }
            a = an;
        }

        // ---- elementwise LSTM cell update; c stays in VGPRs ----
        #pragma unroll
        for (int r = 0; r < 8; ++r) {
            const float gg = tanh_fast(acc[0][r] + bb0);
            const float ii = sigm_fast(acc[1][r] + bb1);
            const float ff = sigm_fast(acc[2][r] + bb2);
            const float oo = sigm_fast(acc[3][r] + bb3);
            const float cn = gg * ii + cfr[r] * ff;
            cfr[r] = cn;
            const float hh = tanh_fast(cn) * oo;
            const int M = m0 + r + hl * 8;       // C/D fragment layout
            const int N = j * 16 + ll;
            hbuf[M * kHid + N] = (__bf16)hh;
            if (t == kSeq - 1) hfin[M * kHid + N] = hh;
        }

        // ---- device-wide barrier: monotonic counter, 16 resident WGs ----
        __threadfence();
        __syncthreads();
        if (tid == 0) {
            __hip_atomic_fetch_add(cnt, 1u, __ATOMIC_RELEASE,
                                   __HIP_MEMORY_SCOPE_AGENT);
            const unsigned tgt = (unsigned)kNWG * (unsigned)(t + 1);
            while (__hip_atomic_load(cnt, __ATOMIC_ACQUIRE,
                                     __HIP_MEMORY_SCOPE_AGENT) < tgt)
                __builtin_amdgcn_s_sleep(2);
        }
        __syncthreads();
    }
}

// ---------------------------------------------------------------------------
// Kernel 3: tiny output projection  out[128,10] = h @ W_ph + b_p  (fp32)
// ---------------------------------------------------------------------------
__global__ void lstm_out(const float* __restrict__ hfin,
                         const float* __restrict__ Wp,
                         const float* __restrict__ bp,
                         float* __restrict__ out)
{
    const int idx = blockIdx.x * blockDim.x + threadIdx.x;
    if (idx >= kBatch * kCls) return;
    const int b = idx / kCls;
    const int c = idx - b * kCls;
    float s = bp[c];
    #pragma unroll 4
    for (int k = 0; k < kHid; ++k)
        s += hfin[b * kHid + k] * Wp[k * kCls + c];
    out[idx] = s;
}

// ---------------------------------------------------------------------------
extern "C" void kernel_launch(void* const* d_in, const int* in_sizes, int n_in,
                              void* d_out, int out_size, void* d_ws, size_t ws_size,
                              hipStream_t stream)
{
    (void)in_sizes; (void)n_in; (void)out_size; (void)ws_size;

    const float* x   = (const float*)d_in[0];
    const float* Wgx = (const float*)d_in[1];
    const float* Wgh = (const float*)d_in[2];
    const float* bg  = (const float*)d_in[3];
    const float* Wix = (const float*)d_in[4];
    const float* Wih = (const float*)d_in[5];
    const float* bi  = (const float*)d_in[6];
    const float* Wfx = (const float*)d_in[7];
    const float* Wfh = (const float*)d_in[8];
    const float* bfp = (const float*)d_in[9];
    const float* Wox = (const float*)d_in[10];
    const float* Woh = (const float*)d_in[11];
    const float* bo  = (const float*)d_in[12];
    const float* Wp  = (const float*)d_in[13];
    const float* bp  = (const float*)d_in[14];

    // workspace layout (~965 KB total)
    char* ws = (char*)d_ws;
    unsigned* cnt  = (unsigned*)(ws + 0);
    float*    bias = (float*)   (ws + 1024);                       //  4 KB
    __bf16*   hbuf = (__bf16*)  (ws + 8192);                       // 64 KB
    float*    hfin = (float*)   (ws + 8192 + 65536);               // 128 KB
    __bf16*   Wt   = (__bf16*)  (ws + 8192 + 65536 + 131072);      // 768 KB

    lstm_pack<<<dim3(kN4H), dim3(kKTot), 0, stream>>>(
        Wgx, Wgh, bg, Wix, Wih, bi, Wfx, Wfh, bfp, Wox, Woh, bo,
        Wt, bias, hbuf, cnt);

    lstm_rec<<<dim3(kNWG), dim3(256), 0, stream>>>(
        x, Wt, bias, hbuf, hfin, cnt);

    lstm_out<<<dim3((kBatch * kCls + 255) / 256), dim3(256), 0, stream>>>(
        hfin, Wp, bp, (float*)d_out);
}
